// RTDETRTransformerv2_70557722739161
// MI455X (gfx1250) — matrix operand order repaired and checked by reading, not compiled, not run on hardware
//
#include <hip/hip_runtime.h>
#include <math.h>

// ---------------- constants (match reference) ----------------
constexpr int kB    = 8;
constexpr int kHD   = 256;
constexpr int kNQ   = 300;
constexpr int kNC   = 80;
constexpr int kNH   = 8;
constexpr int kNL   = 3;
constexpr int kNP   = 4;
constexpr int kNLAY = 6;
constexpr int kDFF  = 1024;
constexpr int kDH   = 32;
constexpr int kPTOT = 8400;          // 6400 + 1600 + 400
constexpr int kRP   = 32;            // padded K for the ref->qpos GEMM
constexpr int kNP16 = 16;            // padded N for the 4-wide box heads

typedef __attribute__((ext_vector_type(16))) _Float16 v16h;
typedef __attribute__((ext_vector_type(8)))  _Float16 v8h;
typedef __attribute__((ext_vector_type(8)))  float    v8f;

// ---------------- elementwise helpers ----------------
__global__ void k_cast16(const float* __restrict__ x, _Float16* __restrict__ y, int n) {
  int i = blockIdx.x * blockDim.x + threadIdx.x;
  if (i < n) y[i] = (_Float16)x[i];
}

__global__ void k_zero16(_Float16* __restrict__ y, int n) {
  int i = blockIdx.x * blockDim.x + threadIdx.x;
  if (i < n) y[i] = (_Float16)0.f;
}

// dst[n,k] (f16, [Ndst,Kdst]) = src[n,k] (f32, [Nsrc,Ksrc]) zero-padded
__global__ void k_castpad(const float* __restrict__ src, _Float16* __restrict__ dst,
                          int Nsrc, int Ksrc, int Ndst, int Kdst) {
  int i = blockIdx.x * blockDim.x + threadIdx.x;
  if (i >= Ndst * Kdst) return;
  int n = i / Kdst, k = i % Kdst;
  dst[i] = (n < Nsrc && k < Ksrc) ? (_Float16)src[n * Ksrc + k] : (_Float16)0.f;
}

// feat [B,C,P] (f32) -> out [B,P,C] (f16), LDS-tiled so both sides coalesce.
// Requires C%16==0 and P%16==0 (true for all levels). Block = 256 threads.
__global__ void k_transpose_cast(const float* __restrict__ f, _Float16* __restrict__ o,
                                 int C, int P) {
  __shared__ float tile[16][17];
  int b  = blockIdx.z;
  int p0 = blockIdx.x * 16;
  int c0 = blockIdx.y * 16;
  int tx = threadIdx.x & 15;
  int ty = threadIdx.x >> 4;
  tile[ty][tx] = f[((size_t)b * C + (c0 + ty)) * P + (p0 + tx)];
  __syncthreads();
  o[((size_t)b * P + (p0 + ty)) * C + (c0 + tx)] = (_Float16)tile[tx][ty];
}

// o = a + b (f32 and/or f16 outputs)
__global__ void k_add2(const float* __restrict__ a, const float* __restrict__ b,
                       float* __restrict__ o32, _Float16* __restrict__ o16, int n) {
  int i = blockIdx.x * blockDim.x + threadIdx.x;
  if (i >= n) return;
  float v = a[i] + b[i];
  if (o32) o32[i] = v;
  if (o16) o16[i] = (_Float16)v;
}

// ---------------- WMMA GEMM: OUT[m,n] = act( sum_k A[m,k]*W[n,k] + bias[n] ) ----------------
// A: f16 [M,K] row-major. W: f16 [N,K] row-major. One wave per 16x16 output tile.
// REQUIRES: M%16==0, N%16==0, K%32==0 (all call sites are padded to satisfy this).
// Output rows remapped: row = (m/rowsPerBatch)*outBatchStride + outRowOff + m%rowsPerBatch

__device__ __forceinline__ v16h load_a_frag(const _Float16* __restrict__ Arow, int kb, int hi) {
  // ISA 7.12.2: 16-bit A 16x32 -> lane holds K = koff..koff+7 and 16+koff..16+koff+7
  v8h a0 = *(const v8h*)(Arow + kb + hi * 8);
  v8h a1 = *(const v8h*)(Arow + kb + 16 + hi * 8);
  v16h a;
  #pragma unroll
  for (int i = 0; i < 8; ++i) { a[i] = a0[i]; a[8 + i] = a1[i]; }
  return a;
}
__device__ __forceinline__ v16h load_b_frag(const _Float16* __restrict__ Wrow, int kb, int hi) {
  // B 32x16 -> lane holds K = koff..koff+7 and 16+koff..16+koff+7  (koff = hi*8), same as A
  v8h b0 = *(const v8h*)(Wrow + kb + hi * 8);
  v8h b1 = *(const v8h*)(Wrow + kb + 16 + hi * 8);
  v16h b;
  #pragma unroll
  for (int i = 0; i < 8; ++i) { b[i] = b0[i]; b[8 + i] = b1[i]; }
  return b;
}

__global__ void k_gemm(const _Float16* __restrict__ A, const _Float16* __restrict__ W,
                       const float* __restrict__ bias,
                       float* __restrict__ o32, _Float16* __restrict__ o16,
                       int M, int N, int K, int ldo,
                       int rowsPerBatch, int outBatchStride, int outRowOff, int act) {
  int wave = (int)((blockIdx.x * blockDim.x + threadIdx.x) >> 5);
  int lane = threadIdx.x & 31;
  int ntiles = N >> 4;
  int mtiles = M >> 4;
  if (wave >= mtiles * ntiles) return;
  int mt = wave / ntiles, nt = wave % ntiles;
  int l16 = lane & 15, hi = lane >> 4;
  const _Float16* Arow = A + (size_t)(mt * 16 + l16) * K;
  const _Float16* Wrow = W + (size_t)(nt * 16 + l16) * K;
  v8f acc = {}, acc2 = {};
  int kb = 0;
  for (; kb + 64 <= K; kb += 64) {
    if (kb + 64 < K) {                      // CDNA5 global_prefetch_b8
      __builtin_prefetch(Arow + kb + 64);
      __builtin_prefetch(Wrow + kb + 64);
    }
    v16h a0 = load_a_frag(Arow, kb, hi);
    v16h b0 = load_b_frag(Wrow, kb, hi);
    v16h a1 = load_a_frag(Arow, kb + 32, hi);
    v16h b1 = load_b_frag(Wrow, kb + 32, hi);
    acc  = __builtin_amdgcn_wmma_f32_16x16x32_f16(false, a0, false, b0, (short)0, acc,
                                                  false, false);
    acc2 = __builtin_amdgcn_wmma_f32_16x16x32_f16(false, a1, false, b1, (short)0, acc2,
                                                  false, false);
  }
  if (kb < K) {                             // K%64==32 tail (e.g. K=32)
    v16h a0 = load_a_frag(Arow, kb, hi);
    v16h b0 = load_b_frag(Wrow, kb, hi);
    acc = __builtin_amdgcn_wmma_f32_16x16x32_f16(false, a0, false, b0, (short)0, acc,
                                                 false, false);
  }
  acc = acc + acc2;
  int nn = nt * 16 + l16;
  float bv = bias ? bias[nn] : 0.f;
  #pragma unroll
  for (int r = 0; r < 8; ++r) {
    int mm = mt * 16 + hi * 8 + r;          // D layout: VGPR r -> M = hi*8+r, N = lane&15
    float v = acc[r] + bv;
    if (act == 1) v = fmaxf(v, 0.f);
    int row = (mm / rowsPerBatch) * outBatchStride + outRowOff + (mm % rowsPerBatch);
    if (o32) o32[(size_t)row * ldo + nn] = v;
    if (o16) o16[(size_t)row * ldo + nn] = (_Float16)v;
  }
}

// ---------------- anchors ----------------
__global__ void k_anchors(float* __restrict__ anc, float* __restrict__ valid) {
  int p = blockIdx.x * blockDim.x + threadIdx.x;
  if (p >= kPTOT) return;
  int lvl, off, w;
  if (p < 6400)      { lvl = 0; off = 0;    w = 80; }
  else if (p < 8000) { lvl = 1; off = 6400; w = 40; }
  else               { lvl = 2; off = 8000; w = 20; }
  int pl = p - off;
  int gy = pl / w, gx = pl % w;
  float wh = 0.05f * (float)(1 << lvl);
  float v[4] = { (gx + 0.5f) / w, (gy + 0.5f) / w, wh, wh };   // h==w per level
  bool ok = true;
  #pragma unroll
  for (int i = 0; i < 4; ++i) ok = ok && (v[i] > 0.01f) && (v[i] < 0.99f);
  #pragma unroll
  for (int i = 0; i < 4; ++i) {
    float lg = logf(v[i] / (1.f - v[i]));
    anc[p * 4 + i] = ok ? lg : 1.0e6f;
  }
  valid[p] = ok ? 1.f : 0.f;
}

// memory = (raw * g + b) * valid   (per level range)
__global__ void k_scale_valid(const float* __restrict__ raw, const float* __restrict__ g,
                              const float* __restrict__ bb, const float* __restrict__ valid,
                              float* __restrict__ mem, _Float16* __restrict__ mem16,
                              int tokOff, int nTok) {
  int i = blockIdx.x * blockDim.x + threadIdx.x;
  int total = kB * nTok * kHD;
  if (i >= total) return;
  int c = i % kHD;
  int p = (i / kHD) % nTok;
  int b = i / (kHD * nTok);
  size_t idx = ((size_t)(b * kPTOT + tokOff + p)) * kHD + c;
  float v = (raw[idx] * g[c] + bb[c]) * valid[tokOff + p];
  mem[idx] = v;
  mem16[idx] = (_Float16)v;
}

// out = LN(x (+resid)) * g + b    -- one 256-thread block per row, C == 256
__global__ void k_addln(const float* __restrict__ x, const float* __restrict__ resid,
                        const float* __restrict__ g, const float* __restrict__ b,
                        float* __restrict__ o32, _Float16* __restrict__ o16, int C) {
  int row = blockIdx.x, c = threadIdx.x;
  __shared__ float red[256];
  float v = x[(size_t)row * C + c];
  if (resid) v += resid[(size_t)row * C + c];
  red[c] = v; __syncthreads();
  for (int s = 128; s > 0; s >>= 1) { if (c < s) red[c] += red[c + s]; __syncthreads(); }
  float mean = red[0] / C; __syncthreads();
  float d = v - mean;
  red[c] = d * d; __syncthreads();
  for (int s = 128; s > 0; s >>= 1) { if (c < s) red[c] += red[c + s]; __syncthreads(); }
  float var = red[0] / C;
  float y = d * rsqrtf(var + 1e-5f) * g[c] + b[c];
  o32[(size_t)row * C + c] = y;
  if (o16) o16[(size_t)row * C + c] = (_Float16)y;
}

__global__ void k_rowmax(const float* __restrict__ s, float* __restrict__ m, int rows, int C) {
  int r = blockIdx.x * blockDim.x + threadIdx.x;
  if (r >= rows) return;
  float mx = -3.4e38f;
  for (int c = 0; c < C; ++c) mx = fmaxf(mx, s[(size_t)r * C + c]);
  m[r] = mx;
}

// descending arg-top-k (NQ of PTOT) per batch; low-index tie break (matches lax.top_k)
__global__ void k_topk(const float* __restrict__ smax, int* __restrict__ topi) {
  int b = blockIdx.x, t = threadIdx.x;
  __shared__ float sc[kPTOT];
  __shared__ float cv[256];
  __shared__ int   ci[256];
  for (int p = t; p < kPTOT; p += 256) sc[p] = smax[b * kPTOT + p];
  __syncthreads();
  for (int it = 0; it < kNQ; ++it) {
    float best = -3.4e38f; int bi = 0x7fffffff;
    for (int p = t; p < kPTOT; p += 256) {
      float v = sc[p];
      if (v > best || (v == best && p < bi)) { best = v; bi = p; }
    }
    cv[t] = best; ci[t] = bi; __syncthreads();
    for (int s = 128; s > 0; s >>= 1) {
      if (t < s) {
        if (cv[t + s] > cv[t] || (cv[t + s] == cv[t] && ci[t + s] < ci[t])) {
          cv[t] = cv[t + s]; ci[t] = ci[t + s];
        }
      }
      __syncthreads();
    }
    if (t == 0) { topi[b * kNQ + it] = ci[0]; sc[ci[0]] = -3.4e38f; }
    __syncthreads();
  }
}

__global__ void k_gather_sel(const float* __restrict__ outmem, const int* __restrict__ topi,
                             float* __restrict__ tgt, _Float16* __restrict__ tgt16) {
  int i = blockIdx.x * blockDim.x + threadIdx.x;
  if (i >= kB * kNQ * kHD) return;
  int c = i % kHD;
  int q = (i / kHD) % kNQ;
  int b = i / (kHD * kNQ);
  float v = outmem[((size_t)(b * kPTOT + topi[b * kNQ + q])) * kHD + c];
  tgt[i] = v; tgt16[i] = (_Float16)v;
}

__global__ void k_gather_anc(const float* __restrict__ anc, const int* __restrict__ topi,
                             float* __restrict__ o) {
  int i = blockIdx.x * blockDim.x + threadIdx.x;
  if (i >= kB * kNQ * 4) return;
  int c = i % 4;
  int q = (i / 4) % kNQ;
  int b = i / (4 * kNQ);
  o[i] = anc[topi[b * kNQ + q] * 4 + c];
}

// ref = sigmoid(delta[.,0:4] + bias + ancsel); delta is [rows, 16] (N-padded GEMM out)
// ref16p is the zero-padded [rows, 32] f16 copy (only cols 0..3 written).
__global__ void k_refinit(const float* __restrict__ delta, const float* __restrict__ bias,
                          const float* __restrict__ ancsel,
                          float* __restrict__ ref, _Float16* __restrict__ ref16p) {
  int i = blockIdx.x * blockDim.x + threadIdx.x;
  if (i >= kB * kNQ * 4) return;
  int c = i % 4, r = i / 4;
  float v = 1.f / (1.f + expf(-(delta[(size_t)r * kNP16 + c] + bias[c] + ancsel[i])));
  ref[i] = v; ref16p[(size_t)r * kRP + c] = (_Float16)v;
}

// ref = sigmoid(delta[.,0:4] + bias + inv_sigmoid(ref))  (in place)
__global__ void k_refupd(const float* __restrict__ delta, const float* __restrict__ bias,
                         float* __restrict__ ref, _Float16* __restrict__ ref16p) {
  int i = blockIdx.x * blockDim.x + threadIdx.x;
  if (i >= kB * kNQ * 4) return;
  int c = i % 4, r = i / 4;
  float x = fminf(fmaxf(ref[i], 0.f), 1.f);
  float inv = logf(fmaxf(x, 1e-5f) / fmaxf(1.f - x, 1e-5f));
  float v = 1.f / (1.f + expf(-(delta[(size_t)r * kNP16 + c] + bias[c] + inv)));
  ref[i] = v; ref16p[(size_t)r * kRP + c] = (_Float16)v;
}

// ---------------- self-attention (8 heads, DH=32, 300 keys) ----------------
__global__ void k_mha(const float* __restrict__ q, const float* __restrict__ k,
                      const float* __restrict__ v, float* __restrict__ o,
                      _Float16* __restrict__ o16) {
  int t = blockIdx.x * blockDim.x + threadIdx.x;
  if (t >= kB * kNH * kNQ) return;
  int qi = t % kNQ;
  int h  = (t / kNQ) % kNH;
  int b  = t / (kNQ * kNH);
  const float scale = 0.17677669529663687f;   // 1/sqrt(32)
  float qv[kDH];
  const float* qp = q + ((size_t)(b * kNQ + qi)) * kHD + h * kDH;
  #pragma unroll
  for (int d = 0; d < kDH; ++d) qv[d] = qp[d] * scale;
  const float* kbase = k + (size_t)b * kNQ * kHD + h * kDH;
  const float* vbase = v + (size_t)b * kNQ * kHD + h * kDH;
  float mx = -3.4e38f;
  for (int j = 0; j < kNQ; ++j) {
    const float* kp = kbase + (size_t)j * kHD;
    float s = 0.f;
    #pragma unroll
    for (int d = 0; d < kDH; ++d) s += qv[d] * kp[d];
    mx = fmaxf(mx, s);
  }
  float l = 0.f;
  float acc[kDH];
  #pragma unroll
  for (int d = 0; d < kDH; ++d) acc[d] = 0.f;
  for (int j = 0; j < kNQ; ++j) {
    const float* kp = kbase + (size_t)j * kHD;
    float s = 0.f;
    #pragma unroll
    for (int d = 0; d < kDH; ++d) s += qv[d] * kp[d];
    float e = expf(s - mx); l += e;
    const float* vp = vbase + (size_t)j * kHD;
    #pragma unroll
    for (int d = 0; d < kDH; ++d) acc[d] += e * vp[d];
  }
  float inv = 1.f / l;
  float*     op  = o   + ((size_t)(b * kNQ + qi)) * kHD + h * kDH;
  _Float16*  op6 = o16 + ((size_t)(b * kNQ + qi)) * kHD + h * kDH;
  #pragma unroll
  for (int d = 0; d < kDH; ++d) { float r = acc[d] * inv; op[d] = r; op6[d] = (_Float16)r; }
}

// softmax over the 12 (NL*NP) weights per (b,q,h); aw layout [B*NQ, NH*12]
__global__ void k_awsoftmax(float* __restrict__ aw) {
  int t = blockIdx.x * blockDim.x + threadIdx.x;
  if (t >= kB * kNQ * kNH) return;
  int h  = t % kNH;
  int bq = t / kNH;
  float* p = aw + (size_t)bq * (kNH * kNL * kNP) + h * (kNL * kNP);
  float mx = -3.4e38f;
  #pragma unroll
  for (int i = 0; i < kNL * kNP; ++i) mx = fmaxf(mx, p[i]);
  float s = 0.f;
  #pragma unroll
  for (int i = 0; i < kNL * kNP; ++i) { float e = expf(p[i] - mx); p[i] = e; s += e; }
  float inv = 1.f / s;
  #pragma unroll
  for (int i = 0; i < kNL * kNP; ++i) p[i] *= inv;
}

// multi-scale deformable attention sampling
__global__ void k_msda(const float* __restrict__ val, const float* __restrict__ off,
                       const float* __restrict__ aw, const float* __restrict__ ref,
                       float* __restrict__ o, _Float16* __restrict__ o16) {
  int t = blockIdx.x * blockDim.x + threadIdx.x;
  if (t >= kB * kNQ * kNH) return;
  int h  = t % kNH;
  int q  = (t / kNH) % kNQ;
  int b  = t / (kNH * kNQ);
  int bq = b * kNQ + q;
  float rx = ref[bq * 4 + 0], ry = ref[bq * 4 + 1];
  float rw = ref[bq * 4 + 2], rh = ref[bq * 4 + 3];
  float acc[kDH];
  #pragma unroll
  for (int d = 0; d < kDH; ++d) acc[d] = 0.f;
  const int offs[3] = {0, 6400, 8000};
  const int wsz[3]  = {80, 40, 20};
  for (int l = 0; l < kNL; ++l) {
    int w = wsz[l];
    const float* vb = val + ((size_t)(b * kPTOT + offs[l])) * kHD + h * kDH;
    for (int p = 0; p < kNP; ++p) {
      int n = (((h * kNL) + l) * kNP + p) * 2;
      float ox = off[(size_t)bq * (kNH * kNL * kNP * 2) + n];
      float oy = off[(size_t)bq * (kNH * kNL * kNP * 2) + n + 1];
      float lx = rx + ox * (1.f / kNP) * rw * 0.5f;
      float ly = ry + oy * (1.f / kNP) * rh * 0.5f;
      float x = lx * w - 0.5f, y = ly * w - 0.5f;   // h==w
      float x0 = floorf(x), y0 = floorf(y);
      float fx = x - x0, fy = y - y0;
      float a = aw[(size_t)bq * (kNH * kNL * kNP) + h * (kNL * kNP) + l * kNP + p];
      #pragma unroll
      for (int dy = 0; dy < 2; ++dy) {
        #pragma unroll
        for (int dx = 0; dx < 2; ++dx) {
          int xi = (int)x0 + dx, yi = (int)y0 + dy;
          if (xi < 0 || xi >= w || yi < 0 || yi >= w) continue;
          float wt = (dx ? fx : 1.f - fx) * (dy ? fy : 1.f - fy) * a;
          const float* g = vb + ((size_t)(yi * w + xi)) * kHD;
          for (int d = 0; d < kDH; ++d) acc[d] += wt * g[d];
        }
      }
    }
  }
  float*    op  = o   + (size_t)bq * kHD + h * kDH;
  _Float16* op6 = o16 + (size_t)bq * kHD + h * kDH;
  #pragma unroll
  for (int d = 0; d < kDH; ++d) { op[d] = acc[d]; op6[d] = (_Float16)acc[d]; }
}

// write ref into out[..., 80:84]
__global__ void k_outref(const float* __restrict__ ref, float* __restrict__ out) {
  int i = blockIdx.x * blockDim.x + threadIdx.x;
  if (i >= kB * kNQ * 4) return;
  int c = i % 4, r = i / 4;
  out[(size_t)r * (kNC + 4) + kNC + c] = ref[i];
}

// ---------------- host orchestration ----------------
static inline int cdiv(long a, int b) { return (int)((a + b - 1) / b); }

extern "C" void kernel_launch(void* const* d_in, const int* in_sizes, int n_in,
                              void* d_out, int out_size, void* d_ws, size_t ws_size,
                              hipStream_t stream) {
  (void)in_sizes; (void)n_in; (void)out_size; (void)ws_size;
  const float* feat[3] = {(const float*)d_in[0], (const float*)d_in[4], (const float*)d_in[8]};
  const float* pw[3]   = {(const float*)d_in[1], (const float*)d_in[5], (const float*)d_in[9]};
  const float* bgv[3]  = {(const float*)d_in[2], (const float*)d_in[6], (const float*)d_in[10]};
  const float* bbv[3]  = {(const float*)d_in[3], (const float*)d_in[7], (const float*)d_in[11]};
  const float* eo_w  = (const float*)d_in[12]; const float* eo_b  = (const float*)d_in[13];
  const float* eln_g = (const float*)d_in[14]; const float* eln_b = (const float*)d_in[15];
  const float* es_w  = (const float*)d_in[16]; const float* es_b  = (const float*)d_in[17];
  const float* eb_w1 = (const float*)d_in[18]; const float* eb_b1 = (const float*)d_in[19];
  const float* eb_w2 = (const float*)d_in[20]; const float* eb_b2 = (const float*)d_in[21];
  const float* eb_w3 = (const float*)d_in[22]; const float* eb_b3 = (const float*)d_in[23];
  const float* qp_w1 = (const float*)d_in[24]; const float* qp_b1 = (const float*)d_in[25];
  const float* qp_w2 = (const float*)d_in[26]; const float* qp_b2 = (const float*)d_in[27];
  const float* d_qkv_w = (const float*)d_in[28]; const float* d_qkv_b = (const float*)d_in[29];
  const float* d_ao_w  = (const float*)d_in[30]; const float* d_ao_b  = (const float*)d_in[31];
  const float* d_ln1_g = (const float*)d_in[32]; const float* d_ln1_b = (const float*)d_in[33];
  const float* d_off_w = (const float*)d_in[34]; const float* d_off_b = (const float*)d_in[35];
  const float* d_aw_w  = (const float*)d_in[36]; const float* d_aw_b  = (const float*)d_in[37];
  const float* d_val_w = (const float*)d_in[38]; const float* d_val_b = (const float*)d_in[39];
  const float* d_out_w = (const float*)d_in[40]; const float* d_out_b = (const float*)d_in[41];
  const float* d_ln2_g = (const float*)d_in[42]; const float* d_ln2_b = (const float*)d_in[43];
  const float* d_f1_w  = (const float*)d_in[44]; const float* d_f1_b  = (const float*)d_in[45];
  const float* d_f2_w  = (const float*)d_in[46]; const float* d_f2_b  = (const float*)d_in[47];
  const float* d_ln3_g = (const float*)d_in[48]; const float* d_ln3_b = (const float*)d_in[49];
  const float* d_sc_w  = (const float*)d_in[50]; const float* d_sc_b  = (const float*)d_in[51];
  const float* d_bb_w1 = (const float*)d_in[52]; const float* d_bb_b1 = (const float*)d_in[53];
  const float* d_bb_w2 = (const float*)d_in[54]; const float* d_bb_b2 = (const float*)d_in[55];
  const float* d_bb_w3 = (const float*)d_in[56]; const float* d_bb_b3 = (const float*)d_in[57];
  float* out = (float*)d_out;

  const int Cl[3]   = {512, 1024, 2048};
  const int Pl[3]   = {6400, 1600, 400};
  const int offl[3] = {0, 6400, 8000};

  // ---- workspace allocator ----
  char* wsbase = (char*)d_ws;
  size_t wsoff = 0;
  auto alloc = [&](size_t bytes) -> void* {
    wsoff = (wsoff + 255) & ~(size_t)255;
    void* p = wsbase + wsoff;
    wsoff += bytes;
    return p;
  };

  // f16 weight copies (box heads N-padded to 16 rows; qp_w1 K-padded to 32)
  _Float16* w_pwh[3];
  for (int l = 0; l < 3; ++l) w_pwh[l] = (_Float16*)alloc((size_t)kHD * Cl[l] * 2);
  _Float16* w_eo   = (_Float16*)alloc((size_t)kHD * kHD * 2);
  _Float16* w_es   = (_Float16*)alloc((size_t)kNC * kHD * 2);
  _Float16* w_eb1  = (_Float16*)alloc((size_t)kHD * kHD * 2);
  _Float16* w_eb2  = (_Float16*)alloc((size_t)kHD * kHD * 2);
  _Float16* w_eb3p = (_Float16*)alloc((size_t)kNP16 * kHD * 2);
  _Float16* w_qp1p = (_Float16*)alloc((size_t)2 * kHD * kRP * 2);
  _Float16* w_qp2  = (_Float16*)alloc((size_t)kHD * 2 * kHD * 2);
  _Float16* w_qkv  = (_Float16*)alloc((size_t)kNLAY * 3 * kHD * kHD * 2);
  _Float16* w_ao   = (_Float16*)alloc((size_t)kNLAY * kHD * kHD * 2);
  _Float16* w_off  = (_Float16*)alloc((size_t)kNLAY * 192 * kHD * 2);
  _Float16* w_aw   = (_Float16*)alloc((size_t)kNLAY * 96 * kHD * 2);
  _Float16* w_val  = (_Float16*)alloc((size_t)kNLAY * kHD * kHD * 2);
  _Float16* w_out  = (_Float16*)alloc((size_t)kNLAY * kHD * kHD * 2);
  _Float16* w_f1   = (_Float16*)alloc((size_t)kNLAY * kDFF * kHD * 2);
  _Float16* w_f2   = (_Float16*)alloc((size_t)kNLAY * kHD * kDFF * 2);
  _Float16* w_bb1  = (_Float16*)alloc((size_t)kNLAY * kHD * kHD * 2);
  _Float16* w_bb2  = (_Float16*)alloc((size_t)kNLAY * kHD * kHD * 2);
  _Float16* w_bb3p = (_Float16*)alloc((size_t)kNLAY * kNP16 * kHD * 2);
  _Float16* w_sc   = (_Float16*)alloc((size_t)kNLAY * kNC * kHD * 2);

  // activations
  const size_t MEMN = (size_t)kB * kPTOT * kHD;       // 17.2M
  _Float16* featT16 = (_Float16*)alloc((size_t)kB * 6400 * 512 * 2);
  float*    mem32   = (float*)alloc(MEMN * 4);
  _Float16* mem16   = (_Float16*)alloc(MEMN * 2);
  float*    bigA    = (float*)alloc(MEMN * 4);        // raw proj -> pre-LN -> scores -> val
  float*    outm32  = (float*)alloc(MEMN * 4);
  _Float16* outm16  = (_Float16*)alloc(MEMN * 2);
  float*    smax    = (float*)alloc((size_t)kB * kPTOT * 4);
  int*      topi    = (int*)alloc((size_t)kB * kNQ * 4);
  float*    anc     = (float*)alloc((size_t)kPTOT * 4 * 4);
  float*    avalid  = (float*)alloc((size_t)kPTOT * 4);
  float*    ancsel  = (float*)alloc((size_t)kB * kNQ * 4 * 4);
  const int QN = kB * kNQ * kHD;                      // 614400
  const int MQ = kB * kNQ;                            // 2400
  float*    tgt32  = (float*)alloc((size_t)QN * 4);
  _Float16* tgt16  = (_Float16*)alloc((size_t)QN * 2);
  float*    qpos32 = (float*)alloc((size_t)QN * 4);
  float*    qin32  = (float*)alloc((size_t)QN * 4);
  _Float16* qin16  = (_Float16*)alloc((size_t)QN * 2);
  _Float16* qc16   = (_Float16*)alloc((size_t)QN * 2);
  float*    qb     = (float*)alloc((size_t)QN * 4);
  float*    kb     = (float*)alloc((size_t)QN * 4);
  float*    vb     = (float*)alloc((size_t)QN * 4);
  float*    att32  = (float*)alloc((size_t)QN * 4);
  _Float16* att16  = (_Float16*)alloc((size_t)QN * 2);
  float*    abuf   = (float*)alloc((size_t)QN * 4);
  _Float16* ffh16  = (_Float16*)alloc((size_t)MQ * kDFF * 2);
  _Float16* h512   = (_Float16*)alloc((size_t)MQ * 512 * 2);
  _Float16* m16a   = (_Float16*)alloc((size_t)QN * 2);
  _Float16* m16b   = (_Float16*)alloc((size_t)QN * 2);
  float*    delta  = (float*)alloc((size_t)MQ * kNP16 * 4);
  float*    offb   = (float*)alloc((size_t)MQ * 192 * 4);
  float*    awb    = (float*)alloc((size_t)MQ * 96 * 4);
  float*    msda32 = (float*)alloc((size_t)QN * 4);
  _Float16* msda16 = (_Float16*)alloc((size_t)QN * 2);
  float*    ref32  = (float*)alloc((size_t)MQ * 4 * 4);
  _Float16* ref16p = (_Float16*)alloc((size_t)MQ * kRP * 2);
  float*    scores = bigA;                             // alias (after LN consumed bigA)

  auto cast = [&](const float* x, _Float16* y, long n) {
    k_cast16<<<cdiv(n, 256), 256, 0, stream>>>(x, y, (int)n);
  };
  auto castpad = [&](const float* x, _Float16* y, int Nsrc, int Ksrc, int Ndst, int Kdst) {
    k_castpad<<<cdiv((long)Ndst * Kdst, 256), 256, 0, stream>>>(x, y, Nsrc, Ksrc, Ndst, Kdst);
  };
  auto gemm = [&](const _Float16* A, const _Float16* W, const float* bias,
                  float* o32, _Float16* o16, int M, int N, int K, int ldo,
                  int rpb, int obs, int oro, int act) {
    int tiles = (M >> 4) * (N >> 4);
    k_gemm<<<cdiv((long)tiles, 4), 128, 0, stream>>>(A, W, bias, o32, o16, M, N, K, ldo,
                                                     rpb, obs, oro, act);
  };

  // ---- cast weights to f16 (with padding where needed) ----
  for (int l = 0; l < 3; ++l) cast(pw[l], w_pwh[l], (long)kHD * Cl[l]);
  cast(eo_w, w_eo, kHD * kHD);      cast(es_w, w_es, kNC * kHD);
  cast(eb_w1, w_eb1, kHD * kHD);    cast(eb_w2, w_eb2, kHD * kHD);
  castpad(eb_w3, w_eb3p, 4, kHD, kNP16, kHD);
  castpad(qp_w1, w_qp1p, 2 * kHD, 4, 2 * kHD, kRP);
  cast(qp_w2, w_qp2, kHD * 2 * kHD);
  cast(d_qkv_w, w_qkv, (long)kNLAY * 3 * kHD * kHD);
  cast(d_ao_w,  w_ao,  (long)kNLAY * kHD * kHD);
  cast(d_off_w, w_off, (long)kNLAY * 192 * kHD);
  cast(d_aw_w,  w_aw,  (long)kNLAY * 96 * kHD);
  cast(d_val_w, w_val, (long)kNLAY * kHD * kHD);
  cast(d_out_w, w_out, (long)kNLAY * kHD * kHD);
  cast(d_f1_w,  w_f1,  (long)kNLAY * kDFF * kHD);
  cast(d_f2_w,  w_f2,  (long)kNLAY * kHD * kDFF);
  cast(d_bb_w1, w_bb1, (long)kNLAY * kHD * kHD);
  cast(d_bb_w2, w_bb2, (long)kNLAY * kHD * kHD);
  for (int i = 0; i < kNLAY; ++i)
    castpad(d_bb_w3 + (size_t)i * 4 * kHD, w_bb3p + (size_t)i * kNP16 * kHD,
            4, kHD, kNP16, kHD);
  cast(d_sc_w,  w_sc,  (long)kNLAY * kNC * kHD);
  k_zero16<<<cdiv((long)MQ * kRP, 256), 256, 0, stream>>>(ref16p, MQ * kRP);

  // ---- encoder ----
  k_anchors<<<cdiv(kPTOT, 256), 256, 0, stream>>>(anc, avalid);
  for (int l = 0; l < 3; ++l) {
    k_transpose_cast<<<dim3(Pl[l] / 16, Cl[l] / 16, kB), 256, 0, stream>>>(
        feat[l], featT16, Cl[l], Pl[l]);
    gemm(featT16, w_pwh[l], nullptr, bigA, nullptr,
         kB * Pl[l], kHD, Cl[l], kHD, Pl[l], kPTOT, offl[l], 0);
    long m = (long)kB * Pl[l] * kHD;
    k_scale_valid<<<cdiv(m, 256), 256, 0, stream>>>(bigA, bgv[l], bbv[l], avalid,
                                                    mem32, mem16, offl[l], Pl[l]);
  }
  const int MTOK = kB * kPTOT;                        // 67200
  gemm(mem16, w_eo, eo_b, bigA, nullptr, MTOK, kHD, kHD, kHD, MTOK, 0, 0, 0);
  k_addln<<<MTOK, 256, 0, stream>>>(bigA, nullptr, eln_g, eln_b, outm32, outm16, kHD);
  gemm(outm16, w_es, es_b, scores, nullptr, MTOK, kNC, kHD, kNC, MTOK, 0, 0, 0);
  k_rowmax<<<cdiv(MTOK, 256), 256, 0, stream>>>(scores, smax, MTOK, kNC);
  k_topk<<<kB, 256, 0, stream>>>(smax, topi);
  k_gather_sel<<<cdiv(QN, 256), 256, 0, stream>>>(outm32, topi, tgt32, tgt16);
  k_gather_anc<<<cdiv(MQ * 4, 256), 256, 0, stream>>>(anc, topi, ancsel);
  gemm(tgt16, w_eb1, eb_b1, nullptr, m16a, MQ, kHD, kHD, kHD, MQ, 0, 0, 1);
  gemm(m16a, w_eb2, eb_b2, nullptr, m16b, MQ, kHD, kHD, kHD, MQ, 0, 0, 1);
  gemm(m16b, w_eb3p, nullptr, delta, nullptr, MQ, kNP16, kHD, kNP16, MQ, 0, 0, 0);
  k_refinit<<<cdiv(MQ * 4, 256), 256, 0, stream>>>(delta, eb_b3, ancsel, ref32, ref16p);

  // ---- decoder layers ----
  for (int i = 0; i < kNLAY; ++i) {
    const _Float16* wq  = w_qkv + (size_t)i * 3 * kHD * kHD;
    const _Float16* wk  = wq + (size_t)kHD * kHD;
    const _Float16* wv  = wq + (size_t)2 * kHD * kHD;
    const float* bqkv = d_qkv_b + (size_t)i * 3 * kHD;

    gemm(ref16p, w_qp1p, qp_b1, nullptr, h512, MQ, 2 * kHD, kRP, 2 * kHD, MQ, 0, 0, 1);
    gemm(h512, w_qp2, qp_b2, qpos32, nullptr, MQ, kHD, 2 * kHD, kHD, MQ, 0, 0, 0);
    k_add2<<<cdiv(QN, 256), 256, 0, stream>>>(tgt32, qpos32, qin32, qin16, QN);
    gemm(qin16, wq, bqkv,            qb, nullptr, MQ, kHD, kHD, kHD, MQ, 0, 0, 0);
    gemm(qin16, wk, bqkv + kHD,      kb, nullptr, MQ, kHD, kHD, kHD, MQ, 0, 0, 0);
    gemm(tgt16, wv, bqkv + 2 * kHD,  vb, nullptr, MQ, kHD, kHD, kHD, MQ, 0, 0, 0);
    k_mha<<<cdiv(kB * kNH * kNQ, 128), 128, 0, stream>>>(qb, kb, vb, att32, att16);
    gemm(att16, w_ao + (size_t)i * kHD * kHD, d_ao_b + (size_t)i * kHD,
         abuf, nullptr, MQ, kHD, kHD, kHD, MQ, 0, 0, 0);
    k_addln<<<MQ, 256, 0, stream>>>(abuf, tgt32, d_ln1_g + (size_t)i * kHD,
                                    d_ln1_b + (size_t)i * kHD, tgt32, tgt16, kHD);
    k_add2<<<cdiv(QN, 256), 256, 0, stream>>>(tgt32, qpos32, nullptr, qc16, QN);
    gemm(qc16, w_off + (size_t)i * 192 * kHD, d_off_b + (size_t)i * 192,
         offb, nullptr, MQ, 192, kHD, 192, MQ, 0, 0, 0);
    gemm(qc16, w_aw + (size_t)i * 96 * kHD, d_aw_b + (size_t)i * 96,
         awb, nullptr, MQ, 96, kHD, 96, MQ, 0, 0, 0);
    k_awsoftmax<<<cdiv(kB * kNQ * kNH, 128), 128, 0, stream>>>(awb);
    gemm(mem16, w_val + (size_t)i * kHD * kHD, d_val_b + (size_t)i * kHD,
         bigA, nullptr, MTOK, kHD, kHD, kHD, MTOK, 0, 0, 0);
    k_msda<<<cdiv(kB * kNQ * kNH, 128), 128, 0, stream>>>(bigA, offb, awb, ref32,
                                                          msda32, msda16);
    gemm(msda16, w_out + (size_t)i * kHD * kHD, d_out_b + (size_t)i * kHD,
         abuf, nullptr, MQ, kHD, kHD, kHD, MQ, 0, 0, 0);
    k_addln<<<MQ, 256, 0, stream>>>(abuf, tgt32, d_ln2_g + (size_t)i * kHD,
                                    d_ln2_b + (size_t)i * kHD, tgt32, tgt16, kHD);
    gemm(tgt16, w_f1 + (size_t)i * kDFF * kHD, d_f1_b + (size_t)i * kDFF,
         nullptr, ffh16, MQ, kDFF, kHD, kDFF, MQ, 0, 0, 1);
    gemm(ffh16, w_f2 + (size_t)i * kHD * kDFF, d_f2_b + (size_t)i * kHD,
         abuf, nullptr, MQ, kHD, kDFF, kHD, MQ, 0, 0, 0);
    k_addln<<<MQ, 256, 0, stream>>>(abuf, tgt32, d_ln3_g + (size_t)i * kHD,
                                    d_ln3_b + (size_t)i * kHD, tgt32, tgt16, kHD);
    gemm(tgt16, w_bb1 + (size_t)i * kHD * kHD, d_bb_b1 + (size_t)i * kHD,
         nullptr, m16a, MQ, kHD, kHD, kHD, MQ, 0, 0, 1);
    gemm(m16a, w_bb2 + (size_t)i * kHD * kHD, d_bb_b2 + (size_t)i * kHD,
         nullptr, m16b, MQ, kHD, kHD, kHD, MQ, 0, 0, 1);
    gemm(m16b, w_bb3p + (size_t)i * kNP16 * kHD, nullptr,
         delta, nullptr, MQ, kNP16, kHD, kNP16, MQ, 0, 0, 0);
    k_refupd<<<cdiv(MQ * 4, 256), 256, 0, stream>>>(delta, d_bb_b3 + (size_t)i * 4,
                                                    ref32, ref16p);
  }

  // ---- final head: logits into out[...,0:80], ref into out[...,80:84] ----
  gemm(tgt16, w_sc + (size_t)(kNLAY - 1) * kNC * kHD, d_sc_b + (size_t)(kNLAY - 1) * kNC,
       out, nullptr, MQ, kNC, kHD, kNC + 4, MQ, 0, 0, 0);
  k_outref<<<cdiv(MQ * 4, 256), 256, 0, stream>>>(ref32, out);
}
